// NequIP_9474697855239
// MI455X (gfx1250) — matrix operand order repaired and checked by reading, not compiled, not run on hardware
//
#include <hip/hip_runtime.h>

#define L0 32
#define NRBF 8
#define HID 64
#define NPATH 6
#define NBLOCKS 5
#define NGRAPHS 16

typedef __attribute__((ext_vector_type(16))) __bf16 v16bf;
typedef __attribute__((ext_vector_type(8)))  __bf16 v8bf;
typedef __attribute__((ext_vector_type(8)))  float  v8f;

#define B_STRIDE 72   // padded bf16 row stride for Wr2^T (144B = 36 dwords, conflict-free)
#define W_STRIDE 64   // bf16 row stride for w^T (edge index dimension)

__device__ __forceinline__ float sigmoidf_(float x) { return 1.0f / (1.0f + __expf(-x)); }
__device__ __forceinline__ float siluf_(float x)    { return x * sigmoidf_(x); }

// ---------------------------------------------------------------------------
// Edge geometry: rbf (E x 8), sh1 (E x 3), sh2 (E x 5)
// ---------------------------------------------------------------------------
__global__ void k_edge_geom(const float* __restrict__ pos,
                            const int* __restrict__ esrc, const int* __restrict__ edst,
                            float* __restrict__ rbf, float* __restrict__ sh1,
                            float* __restrict__ sh2, int E)
{
    int e = blockIdx.x * blockDim.x + threadIdx.x;
    if (e >= E) return;
    int s = esrc[e], t = edst[e];
    float rx = pos[3 * t + 0] - pos[3 * s + 0];
    float ry = pos[3 * t + 1] - pos[3 * s + 1];
    float rz = pos[3 * t + 2] - pos[3 * s + 2];
    float nrm = sqrtf(rx * rx + ry * ry + rz * rz);
    float d = nrm + 1e-9f;
    float inv = 1.0f / d;
    float hx = rx * inv, hy = ry * inv, hz = rz * inv;

    float u = d * (1.0f / 5.0f);                    // RC = 5
    float env = 0.0f;
    if (u < 1.0f) {
        float u2 = u * u, u3 = u2 * u, u6 = u3 * u3, u7 = u6 * u, u8 = u7 * u;
        env = 1.0f - 28.0f * u6 + 48.0f * u7 - 21.0f * u8;   // p = 6
    }
    float pref = sqrtf(2.0f / 5.0f) * env / (d + 1e-9f);
    const float PI_ = 3.14159265358979323846f;
#pragma unroll
    for (int n = 1; n <= NRBF; ++n)
        rbf[(size_t)e * NRBF + (n - 1)] = pref * __sinf((float)n * PI_ * u);

    sh1[(size_t)e * 3 + 0] = hx;
    sh1[(size_t)e * 3 + 1] = hy;
    sh1[(size_t)e * 3 + 2] = hz;

    float c15 = sqrtf(15.0f);
    float inv2s3 = 1.0f / (2.0f * sqrtf(3.0f));
    sh2[(size_t)e * 5 + 0] = c15 * hx * hy;
    sh2[(size_t)e * 5 + 1] = c15 * hy * hz;
    sh2[(size_t)e * 5 + 2] = c15 * (2.0f * hz * hz - hx * hx - hy * hy) * inv2s3;
    sh2[(size_t)e * 5 + 3] = c15 * hz * hx;
    sh2[(size_t)e * 5 + 4] = c15 * (hx * hx - hy * hy) * 0.5f;
}

// ---------------------------------------------------------------------------
// Node feature init: x0 = emb[z], x1 = x2 = 0
// ---------------------------------------------------------------------------
__global__ void k_init_nodes(const int* __restrict__ z, const float* __restrict__ emb,
                             float* __restrict__ x0, float* __restrict__ x1,
                             float* __restrict__ x2, int N)
{
    int i = blockIdx.x * blockDim.x + threadIdx.x;
    if (i < N * L0)     x0[i] = emb[z[i >> 5] * L0 + (i & 31)];
    if (i < N * L0 * 3) x1[i] = 0.0f;
    if (i < N * L0 * 5) x2[i] = 0.0f;
}

__global__ void k_zero_a(float* __restrict__ a0, float* __restrict__ a1,
                         float* __restrict__ a2, int N)
{
    int i = blockIdx.x * blockDim.x + threadIdx.x;
    if (i < N * L0)     a0[i] = 0.0f;
    if (i < N * L0 * 3) a1[i] = 0.0f;
    if (i < N * L0 * 5) a2[i] = 0.0f;
}

__global__ void k_zero_out(float* __restrict__ out, int n)
{
    if ((int)threadIdx.x < n) out[threadIdx.x] = 0.0f;
}

// ---------------------------------------------------------------------------
// Fused edge kernel: h = silu(rbf@Wr1); w = h@Wr2 via v_wmma_f32_16x16x32_bf16;
// messages m0/m1/m2; atomic scatter into a0/a1/a2.
// 128 threads (4 waves), 64 edges per workgroup.
//
// LDS layouts (all fragment accesses contiguous + 16B aligned):
//   hswz  [edge][64]  : A operand, K contiguous per 32-block: [0..7,8..15,16..23,24..31]
//                       -> frag(s,hi) = 16 contiguous bf16 at s*32 + hi*16
//   sWr2T [col][72]   : B operand transposed; frag(s,hi) = K range [s*32+hi*16, +16)
//   wbufT [col][64]   : w transposed (col-major over edges); C writeback = one v8bf
// Total LDS: 2048 + 27648 + 8192 + 24576 = 62464 B
// ---------------------------------------------------------------------------
__global__ __launch_bounds__(128) void k_edge_block(
    const float* __restrict__ rbf, const float* __restrict__ sh1,
    const float* __restrict__ sh2,
    const int* __restrict__ esrc, const int* __restrict__ edst,
    const float* __restrict__ x0, const float* __restrict__ x1,
    const float* __restrict__ x2,
    float* __restrict__ a0, float* __restrict__ a1, float* __restrict__ a2,
    const float* __restrict__ Wr1b, const float* __restrict__ Wr2b, int E)
{
    __shared__ float  sWr1[NRBF * HID];                // 2 KB
    __shared__ __bf16 sWr2T[(NPATH * L0) * B_STRIDE];  // 27648 B
    __shared__ __bf16 hswz[64 * HID];                  // 8192 B
    __shared__ __bf16 wbufT[(NPATH * L0) * W_STRIDE];  // 24576 B

    const int tid = threadIdx.x;
    const int ebase = blockIdx.x * 64;

    for (int i = tid; i < NRBF * HID; i += 128) sWr1[i] = Wr1b[i];
    // transpose Wr2 (HID x 192) -> sWr2T[col][k], coalesced global reads
    for (int i = tid; i < HID * NPATH * L0; i += 128) {
        int k = i / (NPATH * L0);
        int n = i - k * (NPATH * L0);
        sWr2T[n * B_STRIDE + k] = (__bf16)Wr2b[i];
    }
    __syncthreads();

    // ---- phase 1: h = silu(rbf @ Wr1) -> LDS, K-contiguous, vector stores
    {
        int el = tid >> 1;            // local edge 0..63
        int half = tid & 1;           // K (hidden) block: [half*32, half*32+32)
        int e = ebase + el;
        if (e < E) {
            float r[NRBF];
#pragma unroll
            for (int k = 0; k < NRBF; ++k) r[k] = rbf[(size_t)e * NRBF + k];
            float hv[32];
#pragma unroll
            for (int o = 0; o < 32; ++o) {
                int oo = half * 32 + o;
                float acc = 0.0f;
#pragma unroll
                for (int k = 0; k < NRBF; ++k) acc += r[k] * sWr1[k * HID + oo];
                hv[o] = siluf_(acc);
            }
            // contiguous store: positions [0..7]=K0..7, [8..15]=K8..15,
            //                   [16..23]=K16..23, [24..31]=K24..31
            __bf16* orow = &hswz[el * HID + half * 32];
            const int kbase[4] = {0, 8, 16, 24};
#pragma unroll
            for (int g = 0; g < 4; ++g) {
                v8bf pk;
#pragma unroll
                for (int j = 0; j < 8; ++j) pk[j] = (__bf16)hv[kbase[g] + j];
                *(v8bf*)(orow + g * 8) = pk;
            }
        }
    }
    __syncthreads();

    // ---- phase 2: w = h @ Wr2, WMMA bf16 16x16x32, uniform EXEC --------------
    {
        const int lane = tid & 31;
        const int wave = tid >> 5;    // edge tile 0..3 (16 edges each)
        const int row  = lane & 15;
        const int hi   = lane >> 4;

        const __bf16* abase = &hswz[(wave * 16 + row) * HID];
        v16bf afrag0 = *(const v16bf*)(abase +  0 + hi * 16);   // K block 0
        v16bf afrag1 = *(const v16bf*)(abase + 32 + hi * 16);   // K block 1

#pragma unroll
        for (int nt = 0; nt < 12; ++nt) {       // 192 outputs = 12 N-tiles
            int col = nt * 16 + row;
            const __bf16* bbase = &sWr2T[col * B_STRIDE];
            v16bf b0 = *(const v16bf*)(bbase +  0 + hi * 16);
            v16bf b1 = *(const v16bf*)(bbase + 32 + hi * 16);

            v8f acc = {};
            acc = __builtin_amdgcn_wmma_f32_16x16x32_bf16(false, afrag0, false, b0,
                                                          (short)0, acc, false, false);
            acc = __builtin_amdgcn_wmma_f32_16x16x32_bf16(false, afrag1, false, b1,
                                                          (short)0, acc, false, false);
            // C layout: lane holds rows M = hi*8 + j at column col -> one v8bf store
            v8bf cw;
#pragma unroll
            for (int j = 0; j < 8; ++j) cw[j] = (__bf16)acc[j];
            *(v8bf*)&wbufT[col * W_STRIDE + wave * 16 + hi * 8] = cw;
        }
    }
    __syncthreads();

    // ---- phase 3: messages + atomic scatter ----------------------------------
    {
        int el = tid >> 1;
        int sub = tid & 1;            // channels [sub*16, sub*16+16)
        int e = ebase + el;
        if (e < E) {
            int src = esrc[e], dst = edst[e];
            float s1x = sh1[(size_t)e * 3 + 0];
            float s1y = sh1[(size_t)e * 3 + 1];
            float s1z = sh1[(size_t)e * 3 + 2];
            float s2[5];
#pragma unroll
            for (int k = 0; k < 5; ++k) s2[k] = sh2[(size_t)e * 5 + k];

#pragma unroll
            for (int cc = 0; cc < 16; ++cc) {
                int c = sub * 16 + cc;
                float w0 = (float)wbufT[(0 * L0 + c) * W_STRIDE + el];
                float w1 = (float)wbufT[(1 * L0 + c) * W_STRIDE + el];
                float w2 = (float)wbufT[(2 * L0 + c) * W_STRIDE + el];
                float w3 = (float)wbufT[(3 * L0 + c) * W_STRIDE + el];
                float w4 = (float)wbufT[(4 * L0 + c) * W_STRIDE + el];
                float w5 = (float)wbufT[(5 * L0 + c) * W_STRIDE + el];

                float x0j  = x0[(size_t)src * L0 + c];
                float x1j0 = x1[(size_t)src * L0 * 3 + c * 3 + 0];
                float x1j1 = x1[(size_t)src * L0 * 3 + c * 3 + 1];
                float x1j2 = x1[(size_t)src * L0 * 3 + c * 3 + 2];
                float dot1 = x1j0 * s1x + x1j1 * s1y + x1j2 * s1z;

                atomicAdd(&a0[(size_t)dst * L0 + c], w0 * x0j + w1 * dot1);
                atomicAdd(&a1[(size_t)dst * L0 * 3 + c * 3 + 0], w2 * x0j * s1x + w3 * x1j0);
                atomicAdd(&a1[(size_t)dst * L0 * 3 + c * 3 + 1], w2 * x0j * s1y + w3 * x1j1);
                atomicAdd(&a1[(size_t)dst * L0 * 3 + c * 3 + 2], w2 * x0j * s1z + w3 * x1j2);
#pragma unroll
                for (int k = 0; k < 5; ++k) {
                    float x2jk = x2[(size_t)src * L0 * 5 + c * 5 + k];
                    atomicAdd(&a2[(size_t)dst * L0 * 5 + c * 5 + k],
                              w4 * x0j * s2[k] + w5 * x2jk);
                }
            }
        }
    }
}

// ---------------------------------------------------------------------------
// Node update: gates + residual channel mixing. 8 nodes x 32 ch per block.
// ---------------------------------------------------------------------------
__global__ __launch_bounds__(256) void k_node_update(
    const float* __restrict__ a0, const float* __restrict__ a1,
    const float* __restrict__ a2,
    const float* __restrict__ S0b, const float* __restrict__ S1b,
    const float* __restrict__ S2b, const float* __restrict__ G1b,
    const float* __restrict__ G2b,
    float* __restrict__ x0, float* __restrict__ x1, float* __restrict__ x2, int N)
{
    __shared__ float sS0[L0 * L0], sS1[L0 * L0], sS2[L0 * L0];
    __shared__ float sG1[L0 * L0], sG2[L0 * L0];
    __shared__ float sa0[8][L0], sa1[8][L0 * 3], sa2[8][L0 * 5];

    const int tid = threadIdx.x;
    for (int i = tid; i < L0 * L0; i += 256) {
        sS0[i] = S0b[i]; sS1[i] = S1b[i]; sS2[i] = S2b[i];
        sG1[i] = G1b[i]; sG2[i] = G2b[i];
    }
    const int nl = tid >> 5;          // local node 0..7
    const int e  = tid & 31;          // output channel
    const int n  = blockIdx.x * 8 + nl;
    if (n < N) {
        sa0[nl][e] = a0[(size_t)n * L0 + e];
#pragma unroll
        for (int d = 0; d < 3; ++d) sa1[nl][e * 3 + d] = a1[(size_t)n * L0 * 3 + e * 3 + d];
#pragma unroll
        for (int k = 0; k < 5; ++k) sa2[nl][e * 5 + k] = a2[(size_t)n * L0 * 5 + e * 5 + k];
    }
    __syncthreads();

    float t0 = 0.0f, u1 = 0.0f, u2 = 0.0f;
    float y1[3] = {0.0f, 0.0f, 0.0f};
    float y2[5] = {0.0f, 0.0f, 0.0f, 0.0f, 0.0f};
#pragma unroll
    for (int c = 0; c < L0; ++c) {
        float av = sa0[nl][c];
        t0 += av * sS0[c * L0 + e];
        u1 += av * sG1[c * L0 + e];
        u2 += av * sG2[c * L0 + e];
        float s1 = sS1[c * L0 + e];
#pragma unroll
        for (int d = 0; d < 3; ++d) y1[d] += sa1[nl][c * 3 + d] * s1;
        float s2 = sS2[c * L0 + e];
#pragma unroll
        for (int k = 0; k < 5; ++k) y2[k] += sa2[nl][c * 5 + k] * s2;
    }
    if (n < N) {
        float g1 = sigmoidf_(u1), g2 = sigmoidf_(u2);
        x0[(size_t)n * L0 + e] += siluf_(t0);
#pragma unroll
        for (int d = 0; d < 3; ++d) x1[(size_t)n * L0 * 3 + e * 3 + d] += y1[d] * g1;
#pragma unroll
        for (int k = 0; k < 5; ++k) x2[(size_t)n * L0 * 5 + e * 5 + k] += y2[k] * g2;
    }
}

// ---------------------------------------------------------------------------
// Readout: e_atom = silu(x0@oW1)@oW2 + sbias[z]; energy = segsum over batch.
// One node per wave32; shuffle reduction.
// ---------------------------------------------------------------------------
__global__ __launch_bounds__(256) void k_readout(
    const float* __restrict__ x0, const int* __restrict__ z,
    const int* __restrict__ batch,
    const float* __restrict__ oW1, const float* __restrict__ oW2,
    const float* __restrict__ sbias, float* __restrict__ out, int N)
{
    __shared__ float sW1[L0 * L0];
    __shared__ float sx0[8][L0];
    const int tid = threadIdx.x;
    for (int i = tid; i < L0 * L0; i += 256) sW1[i] = oW1[i];
    const int nl = tid >> 5;
    const int e  = tid & 31;
    const int n  = blockIdx.x * 8 + nl;
    if (n < N) sx0[nl][e] = x0[(size_t)n * L0 + e];
    __syncthreads();

    float acc = 0.0f;
#pragma unroll
    for (int c = 0; c < L0; ++c) acc += sx0[nl][c] * sW1[c * L0 + e];
    float v = siluf_(acc) * oW2[e];
#pragma unroll
    for (int off = 16; off > 0; off >>= 1) v += __shfl_xor(v, off, 32);
    if (e == 0 && n < N) atomicAdd(&out[batch[n]], v + sbias[z[n]]);
}

// ---------------------------------------------------------------------------
// Host launcher
// ---------------------------------------------------------------------------
extern "C" void kernel_launch(void* const* d_in, const int* in_sizes, int n_in,
                              void* d_out, int out_size, void* d_ws, size_t ws_size,
                              hipStream_t stream)
{
    (void)n_in; (void)ws_size;
    const int*   z     = (const int*)d_in[0];
    const float* pos   = (const float*)d_in[1];
    const int*   batch = (const int*)d_in[2];
    const int*   esrc  = (const int*)d_in[3];
    const int*   edst  = (const int*)d_in[4];
    const float* emb   = (const float*)d_in[5];
    const float* Wr1   = (const float*)d_in[6];
    const float* Wr2   = (const float*)d_in[7];
    const float* S0    = (const float*)d_in[8];
    const float* S1    = (const float*)d_in[9];
    const float* S2    = (const float*)d_in[10];
    const float* G1    = (const float*)d_in[11];
    const float* G2    = (const float*)d_in[12];
    const float* oW1   = (const float*)d_in[13];
    const float* oW2   = (const float*)d_in[14];
    const float* sbias = (const float*)d_in[15];

    const int N = in_sizes[0];
    const int E = in_sizes[3];

    float* W = (float*)d_ws;
    float* rbf = W;  W += (size_t)E * NRBF;
    float* sh1 = W;  W += (size_t)E * 3;
    float* sh2 = W;  W += (size_t)E * 5;
    float* x0  = W;  W += (size_t)N * L0;
    float* x1  = W;  W += (size_t)N * L0 * 3;
    float* x2  = W;  W += (size_t)N * L0 * 5;
    float* a0  = W;  W += (size_t)N * L0;
    float* a1  = W;  W += (size_t)N * L0 * 3;
    float* a2  = W;  W += (size_t)N * L0 * 5;

    k_zero_out<<<1, 64, 0, stream>>>((float*)d_out, out_size);
    k_edge_geom<<<(E + 255) / 256, 256, 0, stream>>>(pos, esrc, edst, rbf, sh1, sh2, E);
    k_init_nodes<<<(N * L0 * 5 + 255) / 256, 256, 0, stream>>>(z, emb, x0, x1, x2, N);

    for (int b = 0; b < NBLOCKS; ++b) {
        k_zero_a<<<(N * L0 * 5 + 255) / 256, 256, 0, stream>>>(a0, a1, a2, N);
        k_edge_block<<<(E + 63) / 64, 128, 0, stream>>>(
            rbf, sh1, sh2, esrc, edst, x0, x1, x2, a0, a1, a2,
            Wr1 + (size_t)b * NRBF * HID,
            Wr2 + (size_t)b * HID * NPATH * L0, E);
        k_node_update<<<(N + 7) / 8, 256, 0, stream>>>(
            a0, a1, a2,
            S0 + (size_t)b * L0 * L0, S1 + (size_t)b * L0 * L0,
            S2 + (size_t)b * L0 * L0, G1 + (size_t)b * L0 * L0,
            G2 + (size_t)b * L0 * L0,
            x0, x1, x2, N);
    }
    k_readout<<<(N + 7) / 8, 256, 0, stream>>>(x0, z, batch, oW1, oW2, sbias,
                                               (float*)d_out, N);
}
